// MLSTM_6777458393413
// MI455X (gfx1250) — matrix operand order repaired and checked by reading, not compiled, not run on hardware
//
#include <hip/hip_runtime.h>

#define Tn 512
#define Bn 128
#define In 256
#define Hn 512
#define Kn 8
#define On 128
#define KCAT 768   // I + H
#define LDA 776    // padded LDS row stride (bf16 elems) -> conflict-free frag loads
#define NBLK 32

typedef __attribute__((ext_vector_type(16))) __bf16   bf16x16;
typedef __attribute__((ext_vector_type(8)))  float    f32x8;
typedef __attribute__((ext_vector_type(4)))  unsigned uintx4;

union FragU { uintx4 u[2]; bf16x16 v; };

// A-fragment (16x32 bf16, MxK): lane<16 holds row M=lane, K {k0..k0+7, k0+16..k0+23};
// lane>=16 holds row M=lane-16, K {k0+8..k0+15, k0+24..k0+31}.
__device__ __forceinline__ bf16x16 frag_a(const __bf16* rowbase, int k0, int hi) {
  FragU r;
  const uintx4* p = (const uintx4*)(rowbase + k0 + hi * 8);
  r.u[0] = p[0];
  r.u[1] = p[2];   // +16 elements = +32 bytes
  return r.v;
}

// B-fragment (32x16 bf16, KxN): lane<16 holds col N=lane, K {k0..k0+7, k0+16..k0+23};
// lane>=16 holds col N=lane-16, K {k0+8..k0+15, k0+24..k0+31}.
__device__ __forceinline__ bf16x16 frag_b(const __bf16* rowbase, int k0, int hi) {
  FragU r;
  const uintx4* p = (const uintx4*)(rowbase + k0 + hi * 8);
  r.u[0] = p[0];
  r.u[1] = p[2];
  return r.v;
}

__device__ __forceinline__ f32x8 wmma_bf16(bf16x16 a, bf16x16 b, f32x8 c) {
  return __builtin_amdgcn_wmma_f32_16x16x32_bf16(false, a, false, b, (short)0, c,
                                                 false, false);
}

// Async global -> LDS copy (16 bytes per lane), tracked by ASYNCcnt.
__device__ __forceinline__ void async_ld_b128(unsigned lds_byte_addr, const void* gptr) {
  unsigned long long ga = (unsigned long long)(uintptr_t)gptr;
  asm volatile("global_load_async_to_lds_b128 %0, %1, off"
               :: "v"(lds_byte_addr), "v"(ga)
               : "memory");
}
__device__ __forceinline__ void wait_asynccnt0() {
  asm volatile("s_wait_asynccnt 0x0" ::: "memory");
}

__device__ __forceinline__ float sigm(float x)  { return 1.f / (1.f + __expf(-x)); }
__device__ __forceinline__ float tanh_(float x) { return 1.f - 2.f / (__expf(2.f * x) + 1.f); }

__device__ __forceinline__ void grid_barrier(unsigned* cnt, unsigned* gen) {
  __syncthreads();
  if (threadIdx.x == 0) {
    __threadfence();
    unsigned g = __hip_atomic_load(gen, __ATOMIC_RELAXED, __HIP_MEMORY_SCOPE_AGENT);
    unsigned a = __hip_atomic_fetch_add(cnt, 1u, __ATOMIC_ACQ_REL, __HIP_MEMORY_SCOPE_AGENT);
    if (a + 1u == (unsigned)NBLK) {
      __hip_atomic_store(cnt, 0u, __ATOMIC_RELAXED, __HIP_MEMORY_SCOPE_AGENT);
      __hip_atomic_store(gen, g + 1u, __ATOMIC_RELEASE, __HIP_MEMORY_SCOPE_AGENT);
    } else {
      while (__hip_atomic_load(gen, __ATOMIC_ACQUIRE, __HIP_MEMORY_SCOPE_AGENT) == g) {
        __builtin_amdgcn_s_sleep(1);
      }
    }
  }
  __syncthreads();
}

// ---------------- prologue kernels ----------------

__global__ void k_cvt_bf16(const float* __restrict__ src, unsigned short* __restrict__ dst, int n) {
  int i = blockIdx.x * blockDim.x + threadIdx.x;
  if (i < n) ((__bf16*)dst)[i] = (__bf16)src[i];
}

__global__ void k_build_wcat(const float* __restrict__ wih, const float* __restrict__ whh,
                             unsigned short* __restrict__ dst) {
  int i = blockIdx.x * blockDim.x + threadIdx.x;
  if (i >= 4 * Hn * KCAT) return;
  int row = i / KCAT, c = i - row * KCAT;
  float v = (c < In) ? wih[row * In + c] : whh[row * Hn + (c - In)];
  ((__bf16*)dst)[i] = (__bf16)v;
}

__global__ void k_build_wd(const float* __restrict__ wd, const float* __restrict__ ud,
                           unsigned short* __restrict__ dst) {
  int i = blockIdx.x * blockDim.x + threadIdx.x;
  if (i >= 16 * KCAT) return;
  int row = i / KCAT, c = i - row * KCAT;
  float v = 0.f;
  if (row < Kn) v = (c < In) ? wd[row * In + c] : ud[row * Hn + (c - In)];
  ((__bf16*)dst)[i] = (__bf16)v;
}

__global__ void k_init(unsigned short* hb, float* P, unsigned* bar) {
  int i = blockIdx.x * blockDim.x + threadIdx.x;
  if (i < Bn * Hn) ((__bf16*)hb)[i] = (__bf16)0.f;          // h0 = 0 (buffer 0)
  if (i < Kn * Bn * Hn) P[i] = 0.f;                          // D0 = 0 (ring buffer)
  if (i < 2) bar[i] = 0u;                                    // barrier count/gen
}

// ---------------- main persistent kernel ----------------

__global__ __launch_bounds__(256)
void mlstm_main(const unsigned short* __restrict__ Xb_,
                const unsigned short* __restrict__ Wcat_,
                const unsigned short* __restrict__ Wd_,
                const unsigned short* __restrict__ Wout_,
                unsigned short* hb_,
                const float* __restrict__ bias,
                const float* __restrict__ b_d,
                const float* __restrict__ b_out,
                float* out, float* hT, float* cT, float* P,
                unsigned* bar_cnt, unsigned* bar_gen)
{
  const __bf16* Xb   = (const __bf16*)Xb_;
  const __bf16* Wcat = (const __bf16*)Wcat_;
  const __bf16* Wd   = (const __bf16*)Wd_;
  const __bf16* Wout = (const __bf16*)Wout_;
  __bf16*       hb   = (__bf16*)hb_;

  __shared__ __align__(16) __bf16 sA[16 * LDA];  // [x_t | h_t] rows for this m-tile
  const unsigned sA_base = (unsigned)(uintptr_t)sA;

  const int tid  = threadIdx.x;
  const int lane = tid & 31;
  const int wv   = tid >> 5;               // wave 0..7 in workgroup
  const int wg   = blockIdx.x;             // 0..31
  const int lmod = lane & 15;
  const int hi   = (lane >> 4) & 1;
  const int hi8  = hi << 3;
  const int m0   = (wg >> 2) * 16;                    // batch-tile base (8 tiles)
  const int n0   = (((wg & 3) << 3) + wv) * 16;       // H-tile base (32 tiles)
  const int col  = n0 + lmod;                         // this lane's H column
  const bool doY = ((wg & 3) == 0);                   // 64 waves own the y GEMM
  const int ycol = wv * 16 + lmod;

  const __bf16* arow   = sA + lmod * LDA;
  const __bf16* wrow_i = Wcat + (size_t)(0 * Hn + n0 + lmod) * KCAT;
  const __bf16* wrow_f = Wcat + (size_t)(1 * Hn + n0 + lmod) * KCAT;
  const __bf16* wrow_g = Wcat + (size_t)(2 * Hn + n0 + lmod) * KCAT;
  const __bf16* wrow_o = Wcat + (size_t)(3 * Hn + n0 + lmod) * KCAT;
  const __bf16* wrow_d = Wd   + (size_t)lmod * KCAT;
  const __bf16* wrow_y = Wout + (size_t)(wv * 16 + lmod) * Hn;

  const float bi  = bias[0 * Hn + col];
  const float bf_ = bias[1 * Hn + col];
  const float bg  = bias[2 * Hn + col];
  const float bo  = bias[3 * Hn + col];
  const float bd  = (lmod < Kn) ? b_d[lmod] : 0.f;
  const float byo = b_out[ycol];

  for (int t = 0; t <= Tn; ++t) {
    grid_barrier(bar_cnt, bar_gen);   // previous step's h/P writes visible

    // ---- stage A = [x_t | h_t] rows m0..m0+15 into LDS via async DMA ----
    {
      const __bf16* hsrc = hb + (size_t)(t & 1) * Bn * Hn;
      if (t < Tn) {
        // x part: 16 rows x 256 bf16 = 16 x 32 chunks of 16B
        for (int idx = tid; idx < 16 * 32; idx += 256) {
          int r = idx >> 5, c = (idx & 31) * 8;
          async_ld_b128(sA_base + (unsigned)(r * LDA + c) * 2,
                        Xb + ((size_t)t * Bn + m0 + r) * In + c);
        }
      }
      // h part: 16 rows x 512 bf16 = 16 x 64 chunks of 16B
      for (int idx = tid; idx < 16 * 64; idx += 256) {
        int r = idx >> 6, c = (idx & 63) * 8;
        async_ld_b128(sA_base + (unsigned)(r * LDA + In + c) * 2,
                      hsrc + (size_t)(m0 + r) * Hn + c);
      }
      wait_asynccnt0();
    }
    __syncthreads();

    // prefetch next timestep's x panel into cache while we compute
    if (t + 1 < Tn) {
      const __bf16* nx = Xb + ((size_t)(t + 1) * Bn + m0 + (tid >> 4)) * In + (tid & 15) * 16;
      __builtin_prefetch(nx, 0, 1);
    }

    // ---- y_{t-1} = h_t @ W_out^T + b_out (overlaps with gate GEMM) ----
    if (t > 0 && doY) {
      f32x8 y = {};
      for (int ks = 0; ks < Hn / 32; ++ks) {
        int k0 = ks * 32;
        bf16x16 a = frag_a(arow + In, k0, hi);   // h part of staged A
        bf16x16 b = frag_b(wrow_y, k0, hi);
        y = wmma_bf16(a, b, y);
      }
      for (int j = 0; j < 8; ++j) {
        int bb = m0 + j + hi8;
        out[((size_t)(t - 1) * Bn + bb) * On + ycol] = y[j] + byo;
      }
    }

    if (t < Tn) {
      // ---- fused gate + alpha-logit GEMM over K = 768 ----
      f32x8 gi = {}, gf = {}, gg = {}, go = {}, ga = {};
      for (int ks = 0; ks < KCAT / 32; ++ks) {
        int k0 = ks * 32;
        bf16x16 a = frag_a(arow, k0, hi);
        gi = wmma_bf16(a, frag_b(wrow_i, k0, hi), gi);
        gf = wmma_bf16(a, frag_b(wrow_f, k0, hi), gf);
        gg = wmma_bf16(a, frag_b(wrow_g, k0, hi), gg);
        go = wmma_bf16(a, frag_b(wrow_o, k0, hi), go);
        ga = wmma_bf16(a, frag_b(wrow_d, k0, hi), ga);
      }

      // ---- softmax over K=8 (cols 0..7 of alpha tile; 8..15 are zero pad) ----
      float alpha[8];
      for (int j = 0; j < 8; ++j) {
        float v = ga[j] + bd;
        float m = v;
        m = fmaxf(m, __shfl_xor(m, 1, 32));
        m = fmaxf(m, __shfl_xor(m, 2, 32));
        m = fmaxf(m, __shfl_xor(m, 4, 32));
        float e = __expf(v - m);
        float s = e;
        s += __shfl_xor(s, 1, 32);
        s += __shfl_xor(s, 2, 32);
        s += __shfl_xor(s, 4, 32);
        alpha[j] = e / s;
      }

      // ---- c_hist = sum_k alpha[b,k] * D[k,b,h]; D[k] lives in P[(t+k)&7] ----
      f32x8 ch = {};
      const int sbase = lane & 16;
      for (int k = 0; k < Kn; ++k) {
        const float* Pk = P + (size_t)((t + k) & 7) * Bn * Hn;
        for (int j = 0; j < 8; ++j) {
          float coeff = __shfl(alpha[j], sbase + k, 32);
          ch[j] = fmaf(coeff, Pk[(size_t)(m0 + j + hi8) * Hn + col], ch[j]);
        }
      }

      // ---- gate nonlinearities, state update, ring write ----
      float*  Pw = P  + (size_t)(t & 7) * Bn * Hn;      // overwrite oldest slot
      __bf16* hw = hb + (size_t)((t + 1) & 1) * Bn * Hn;
      for (int j = 0; j < 8; ++j) {
        int bb = m0 + j + hi8;
        float iv = sigm(gi[j] + bi);
        float fv = sigm(gf[j] + bf_);
        float gv = tanh_(gg[j] + bg);
        float ov = sigm(go[j] + bo);
        float cn = fv * ch[j] + iv * gv;
        float hn = ov * tanh_(cn);
        Pw[(size_t)bb * Hn + col] = cn;
        hw[(size_t)bb * Hn + col] = (__bf16)hn;
        if (t == Tn - 1) {
          hT[bb * Hn + col] = hn;
          cT[bb * Hn + col] = cn;
        }
      }
    }
  }
}

// ---------------- host ----------------

extern "C" void kernel_launch(void* const* d_in, const int* in_sizes, int n_in,
                              void* d_out, int out_size, void* d_ws, size_t ws_size,
                              hipStream_t stream) {
  const float* inp   = (const float*)d_in[0];  // [T,B,I]
  const float* W_ih  = (const float*)d_in[1];  // [4H,I]
  const float* W_hh  = (const float*)d_in[2];  // [4H,H]
  const float* b     = (const float*)d_in[3];  // [4H]
  const float* W_d   = (const float*)d_in[4];  // [K,I]
  const float* U_d   = (const float*)d_in[5];  // [K,H]
  const float* b_d   = (const float*)d_in[6];  // [K]
  const float* W_out = (const float*)d_in[7];  // [O,H]
  const float* b_out = (const float*)d_in[8];  // [O]

  char* ws = (char*)d_ws;
  size_t off = 0;
  unsigned short* Xb   = (unsigned short*)(ws + off); off += (size_t)Tn * Bn * In * 2;   // 33.5 MB
  unsigned short* Wcat = (unsigned short*)(ws + off); off += (size_t)4 * Hn * KCAT * 2;  // 3 MB
  unsigned short* Wdp  = (unsigned short*)(ws + off); off += (size_t)16 * KCAT * 2;
  unsigned short* Woutb= (unsigned short*)(ws + off); off += (size_t)On * Hn * 2;
  unsigned short* hb   = (unsigned short*)(ws + off); off += (size_t)2 * Bn * Hn * 2;
  unsigned*       bar  = (unsigned*)(ws + off);       off += 64;

  float* out = (float*)d_out;
  float* hT  = out + (size_t)Tn * Bn * On;           // + 8,388,608
  float* cT  = hT + (size_t)Bn * Hn;
  float* P   = cT + (size_t)Bn * Hn;                 // DT region doubles as ring buffer

  const int thr = 256;
  k_cvt_bf16 <<<(Tn * Bn * In + thr - 1) / thr, thr, 0, stream>>>(inp, Xb, Tn * Bn * In);
  k_build_wcat<<<(4 * Hn * KCAT + thr - 1) / thr, thr, 0, stream>>>(W_ih, W_hh, Wcat);
  k_build_wd <<<(16 * KCAT + thr - 1) / thr, thr, 0, stream>>>(W_d, U_d, Wdp);
  k_cvt_bf16 <<<(On * Hn + thr - 1) / thr, thr, 0, stream>>>(W_out, Woutb, On * Hn);
  k_init     <<<(Kn * Bn * Hn + thr - 1) / thr, thr, 0, stream>>>(hb, P, bar);

  mlstm_main<<<NBLK, thr, 0, stream>>>(Xb, Wcat, Wdp, Woutb, hb, b, b_d, b_out,
                                       out, hT, cT, P, bar, bar + 1);
}
